// EdgeMessageLayer_75831942578740
// MI455X (gfx1250) — matrix operand order repaired and checked by reading, not compiled, not run on hardware
//
#include <hip/hip_runtime.h>
#include <hip/hip_bf16.h>

// ---------------------------------------------------------------------------
// EdgeMessageLayer for MI455X (gfx1250, wave32, WMMA bf16 -> f32 accumulate)
//   N=100000 nodes, E=1600000 edges, H=64, ED=32, block_in=160
// ---------------------------------------------------------------------------

typedef __bf16 bf16_t;
typedef __attribute__((ext_vector_type(16))) __bf16 v16bf;
typedef __attribute__((ext_vector_type(8)))  __bf16 v8bf;
typedef __attribute__((ext_vector_type(8)))  float  v8f;

#define HDIM 64
#define WAVES_PER_BLOCK 8
#define TILE_M 16   // rows (edges / nodes) per wave

union V16U { v16bf v; v8bf h[2]; uint4 q[2]; };

static __device__ __forceinline__ v8bf cvt8(v8f f) {
    // native packed f32 -> bf16 conversion path
    return __builtin_convertvector(f, v8bf);
}

static __device__ __forceinline__ v8f wmma_bf16(v16bf a, v16bf b, v8f c) {
    // D(f32 16x16) = A(bf16 16x32) * B(bf16 32x16) + C
    return __builtin_amdgcn_wmma_f32_16x16x32_bf16(
        /*neg_a=*/false, a, /*neg_b=*/false, b,
        /*c_mod=*/(short)0, c, /*reuse_a=*/false, /*reuse_b=*/false);
}

// load 8 consecutive f32 and convert to 8 bf16 (one half of an A operand)
static __device__ __forceinline__ v8bf load8_bf(const float* __restrict__ p) {
    float4 q0 = *(const float4*)(p);
    float4 q1 = *(const float4*)(p + 4);
    v8f f;
    f[0] = q0.x; f[1] = q0.y; f[2] = q0.z; f[3] = q0.w;
    f[4] = q1.x; f[5] = q1.y; f[6] = q1.z; f[7] = q1.w;
    return cvt8(f);
}

// ---------------------------------------------------------------------------
// Zero-fill (agg accumulator lives in d_out)
// ---------------------------------------------------------------------------
__global__ void zero_f4(float4* __restrict__ p, long long n4) {
    long long i = (long long)blockIdx.x * blockDim.x + threadIdx.x;
    if (i < n4) p[i] = make_float4(0.f, 0.f, 0.f, 0.f);
}

// ---------------------------------------------------------------------------
// Pack a [K,64] f32 weight matrix into WMMA B-operand bf16 layout.
//   P[((kt*NT+nt)*32 + lane)*16 + e] = W[(kt*32 + 8*(lane>>4) + (e&7) + 16*(e>>3))*64 + nt*16 + (lane&15)]
// ---------------------------------------------------------------------------
__global__ void pack_w(const float* __restrict__ W, bf16_t* __restrict__ P,
                       int ktiles, int ntiles) {
    int total = ktiles * ntiles * 512;
    for (int i = blockIdx.x * blockDim.x + threadIdx.x; i < total;
         i += gridDim.x * blockDim.x) {
        int e    = i & 15;
        int lane = (i >> 4) & 31;
        int tile = i >> 9;
        int nt   = tile % ntiles;
        int kt   = tile / ntiles;
        int K    = kt * 32 + 8 * (lane >> 4) + (e & 7) + 16 * (e >> 3);
        int Ncol = nt * 16 + (lane & 15);
        P[i] = (bf16_t)W[K * HDIM + Ncol];
    }
}

// ---------------------------------------------------------------------------
// Edge kernel: messages = MLP(concat(x[src], x[dst], edge_attr)); atomic
// scatter-add into agg[dst]. 8 waves/block, 16 edges per wave.
// ---------------------------------------------------------------------------
__global__ __launch_bounds__(256) void edge_kernel(
    const float* __restrict__ x, const long long* __restrict__ ei,
    const float* __restrict__ ea,
    const bf16_t* __restrict__ w1p, const float* __restrict__ b1,
    const bf16_t* __restrict__ w2p, const float* __restrict__ b2,
    float* __restrict__ agg, int Ecnt)
{
    __shared__ bf16_t hlds[WAVES_PER_BLOCK][TILE_M][HDIM];
    __shared__ int    dstl[WAVES_PER_BLOCK][TILE_M];

    const int lane = threadIdx.x & 31;
    const int wave = threadIdx.x >> 5;
    const int m    = lane & 15;   // row within tile
    const int kh   = lane >> 4;   // K-half selector (A/B layouts)

    const long long ebase = ((long long)blockIdx.x * WAVES_PER_BLOCK + wave) * TILE_M;
    long long e = ebase + m;
    if (e >= Ecnt) e = Ecnt - 1;
    const long long srcI = ei[e];
    const long long dstI = ei[(long long)Ecnt + e];
    dstl[wave][m] = (int)dstI;   // lanes m and m+16 write identical values

    // ---- Build A tiles: msg_in row = [x[src](64) | x[dst](64) | ea(32)] ----
    // A(16x32 bf16): lane m(=lane&15), elements 0..7 -> K = kt*32+kh*8+{0..7},
    //                elements 8..15 -> +16. Chunks never cross concat bounds.
    v16bf a[5];
    #pragma unroll
    for (int kt = 0; kt < 5; ++kt) {
        V16U u;
        #pragma unroll
        for (int half = 0; half < 2; ++half) {
            const int fr = kh * 8 + half * 16;   // 0..24
            const float* p;
            if (kt < 2)      p = x  + srcI * HDIM + (kt * 32 + fr);
            else if (kt < 4) p = x  + dstI * HDIM + ((kt - 2) * 32 + fr);
            else             p = ea + e * 32 + fr;
            u.h[half] = load8_bf(p);
        }
        a[kt] = u.v;
    }

    // ---- Layer 1: [16,160] @ [160,64] + b1, ReLU -> LDS (bf16) ----
    #pragma unroll
    for (int nt = 0; nt < 4; ++nt) {
        v8f c;
        const float bias = b1[nt * 16 + m];
        #pragma unroll
        for (int r = 0; r < 8; ++r) c[r] = bias;
        #pragma unroll
        for (int kt = 0; kt < 5; ++kt) {
            v16bf bw = *(const v16bf*)(w1p + (((kt << 2) + nt) << 9) + (lane << 4));
            c = wmma_bf16(a[kt], bw, c);
        }
        #pragma unroll
        for (int r = 0; r < 8; ++r) c[r] = fmaxf(c[r], 0.f);
        v8bf cb = cvt8(c);
        #pragma unroll
        for (int r = 0; r < 8; ++r) {
            hlds[wave][r + kh * 8][nt * 16 + m] = cb[r];
        }
    }
    __syncthreads();

    // ---- Re-layout hidden C->A via LDS ----
    v16bf ha[2];
    #pragma unroll
    for (int kt = 0; kt < 2; ++kt) {
        V16U u;
        const int K0 = kt * 32 + kh * 8;
        u.q[0] = *(const uint4*)(&hlds[wave][m][K0]);
        u.q[1] = *(const uint4*)(&hlds[wave][m][K0 + 16]);
        ha[kt] = u.v;
    }

    // ---- Layer 2: [16,64] @ [64,64] + b2, then scatter-add by dst ----
    #pragma unroll
    for (int nt = 0; nt < 4; ++nt) {
        v8f c;
        const float bias = b2[nt * 16 + m];
        #pragma unroll
        for (int r = 0; r < 8; ++r) c[r] = bias;
        #pragma unroll
        for (int kt = 0; kt < 2; ++kt) {
            v16bf bw = *(const v16bf*)(w2p + (((kt << 2) + nt) << 9) + (lane << 4));
            c = wmma_bf16(ha[kt], bw, c);
        }
        #pragma unroll
        for (int r = 0; r < 8; ++r) {
            const int M = r + kh * 8;
            if (ebase + M < Ecnt) {
                atomicAdd(agg + (long long)dstl[wave][M] * HDIM + nt * 16 + m, c[r]);
            }
        }
    }
}

// ---------------------------------------------------------------------------
// Node kernel: out = LN(x + SiLU(MLP(concat(x, agg)))). agg aliases out:
// each workgroup only reads/writes its own node rows, ordered by barriers.
// ---------------------------------------------------------------------------
__global__ __launch_bounds__(256) void node_kernel(
    const float* __restrict__ x, const float* __restrict__ agg,
    const bf16_t* __restrict__ w1p, const float* __restrict__ b1,
    const bf16_t* __restrict__ w2p, const float* __restrict__ b2,
    const float* __restrict__ lng, const float* __restrict__ lnb,
    float* __restrict__ out, int Ncnt)
{
    __shared__ bf16_t hlds[WAVES_PER_BLOCK][TILE_M][HDIM];
    __shared__ float  h2[WAVES_PER_BLOCK][TILE_M][HDIM];

    const int lane = threadIdx.x & 31;
    const int wave = threadIdx.x >> 5;
    const int m    = lane & 15;
    const int kh   = lane >> 4;

    const long long nb   = ((long long)blockIdx.x * WAVES_PER_BLOCK + wave) * TILE_M;
    const long long node = nb + m;
    const bool vrow = (node < Ncnt);

    // ---- A tiles: upd_in row = [x(64) | agg(64)] ----
    v16bf a[4];
    #pragma unroll
    for (int kt = 0; kt < 4; ++kt) {
        V16U u;
        #pragma unroll
        for (int half = 0; half < 2; ++half) {
            const int fr = kh * 8 + half * 16;
            const float* p;
            if (!vrow)       p = x;  // never read out-aliased agg for dead rows
            else if (kt < 2) p = x   + node * HDIM + (kt * 32 + fr);
            else             p = agg + node * HDIM + ((kt - 2) * 32 + fr);
            u.h[half] = load8_bf(p);
        }
        a[kt] = u.v;
    }

    // ---- upd layer 1 ----
    #pragma unroll
    for (int nt = 0; nt < 4; ++nt) {
        v8f c;
        const float bias = b1[nt * 16 + m];
        #pragma unroll
        for (int r = 0; r < 8; ++r) c[r] = bias;
        #pragma unroll
        for (int kt = 0; kt < 4; ++kt) {
            v16bf bw = *(const v16bf*)(w1p + (((kt << 2) + nt) << 9) + (lane << 4));
            c = wmma_bf16(a[kt], bw, c);
        }
        #pragma unroll
        for (int r = 0; r < 8; ++r) c[r] = fmaxf(c[r], 0.f);
        v8bf cb = cvt8(c);
        #pragma unroll
        for (int r = 0; r < 8; ++r) {
            hlds[wave][r + kh * 8][nt * 16 + m] = cb[r];
        }
    }
    __syncthreads();

    v16bf ha[2];
    #pragma unroll
    for (int kt = 0; kt < 2; ++kt) {
        V16U u;
        const int K0 = kt * 32 + kh * 8;
        u.q[0] = *(const uint4*)(&hlds[wave][m][K0]);
        u.q[1] = *(const uint4*)(&hlds[wave][m][K0 + 16]);
        ha[kt] = u.v;
    }

    // ---- upd layer 2, residual + SiLU into LDS (f32) ----
    #pragma unroll
    for (int nt = 0; nt < 4; ++nt) {
        v8f c;
        const float bias = b2[nt * 16 + m];
        #pragma unroll
        for (int r = 0; r < 8; ++r) c[r] = bias;
        #pragma unroll
        for (int kt = 0; kt < 2; ++kt) {
            v16bf bw = *(const v16bf*)(w2p + (((kt << 2) + nt) << 9) + (lane << 4));
            c = wmma_bf16(ha[kt], bw, c);
        }
        #pragma unroll
        for (int r = 0; r < 8; ++r) {
            const int M = r + kh * 8;
            const long long nr = nb + M;
            float xv = 0.f;
            if (nr < Ncnt) xv = x[nr * HDIM + nt * 16 + m];
            const float o = c[r];
            const float s = o / (1.f + __expf(-o));  // SiLU
            h2[wave][M][nt * 16 + m] = xv + s;
        }
    }
    __syncthreads();

    // ---- LayerNorm over H=64: 2 lanes per row, shfl_xor(16) reduction ----
    float sum = 0.f, ss = 0.f;
    const float* hr = &h2[wave][m][kh * 32];
    #pragma unroll
    for (int j = 0; j < 8; ++j) {
        float4 q = *(const float4*)(hr + j * 4);
        sum += q.x + q.y + q.z + q.w;
        ss  += q.x * q.x + q.y * q.y + q.z * q.z + q.w * q.w;
    }
    sum += __shfl_xor(sum, 16, 32);
    ss  += __shfl_xor(ss, 16, 32);
    const float mu  = sum * (1.f / 64.f);
    const float var = ss * (1.f / 64.f) - mu * mu;
    const float rs  = rsqrtf(var + 1e-5f);
    if (vrow) {
        #pragma unroll
        for (int j = 0; j < 8; ++j) {
            const int f = kh * 32 + j * 4;
            float4 q  = *(const float4*)(hr + j * 4);
            float4 gg = *(const float4*)(lng + f);
            float4 bb = *(const float4*)(lnb + f);
            float4 o;
            o.x = (q.x - mu) * rs * gg.x + bb.x;
            o.y = (q.y - mu) * rs * gg.y + bb.y;
            o.z = (q.z - mu) * rs * gg.z + bb.z;
            o.w = (q.w - mu) * rs * gg.w + bb.w;
            *(float4*)(out + node * HDIM + f) = o;
        }
    }
}

// ---------------------------------------------------------------------------
extern "C" void kernel_launch(void* const* d_in, const int* in_sizes, int n_in,
                              void* d_out, int out_size, void* d_ws, size_t ws_size,
                              hipStream_t stream) {
    const float*     x       = (const float*)d_in[0];
    const long long* ei      = (const long long*)d_in[1];
    const float*     ea      = (const float*)d_in[2];
    const float*     msg_w1  = (const float*)d_in[3];
    const float*     msg_b1  = (const float*)d_in[4];
    const float*     msg_w2  = (const float*)d_in[5];
    const float*     msg_b2  = (const float*)d_in[6];
    const float*     upd_w1  = (const float*)d_in[7];
    const float*     upd_b1  = (const float*)d_in[8];
    const float*     upd_w2  = (const float*)d_in[9];
    const float*     upd_b2  = (const float*)d_in[10];
    const float*     ln_g    = (const float*)d_in[11];
    const float*     ln_b    = (const float*)d_in[12];

    const int E_ = in_sizes[2] / 32;   // edge_attr is [E, 32]
    const int N_ = in_sizes[0] / HDIM; // x is [N, 64]

    float* agg_out = (float*)d_out;    // 25.6 MB: agg accumulator then output

    // packed bf16 weights in workspace
    bf16_t* wsb     = (bf16_t*)d_ws;
    bf16_t* pw_msg1 = wsb;             // 5*4*512 = 10240
    bf16_t* pw_msg2 = wsb + 10240;     // 2*4*512 =  4096
    bf16_t* pw_upd1 = wsb + 14336;     // 4*4*512 =  8192
    bf16_t* pw_upd2 = wsb + 22528;     // 2*4*512 =  4096

    // 1) zero the aggregation buffer (d_out)
    {
        long long n4 = (long long)N_ * HDIM / 4;
        int blocks = (int)((n4 + 255) / 256);
        zero_f4<<<blocks, 256, 0, stream>>>((float4*)d_out, n4);
    }
    // 2) pack weights to bf16 WMMA B-layout
    pack_w<<<40, 256, 0, stream>>>(msg_w1, pw_msg1, 5, 4);
    pack_w<<<16, 256, 0, stream>>>(msg_w2, pw_msg2, 2, 4);
    pack_w<<<32, 256, 0, stream>>>(upd_w1, pw_upd1, 4, 4);
    pack_w<<<16, 256, 0, stream>>>(upd_w2, pw_upd2, 2, 4);
    // 3) edge messages + scatter-add
    {
        int blocks = (E_ + WAVES_PER_BLOCK * TILE_M - 1) / (WAVES_PER_BLOCK * TILE_M);
        edge_kernel<<<blocks, 256, 0, stream>>>(x, ei, ea, pw_msg1, msg_b1,
                                                pw_msg2, msg_b2, agg_out, E_);
    }
    // 4) node update + residual/SiLU/LayerNorm
    {
        int blocks = (N_ + WAVES_PER_BLOCK * TILE_M - 1) / (WAVES_PER_BLOCK * TILE_M);
        node_kernel<<<blocks, 256, 0, stream>>>(x, agg_out, pw_upd1, upd_b1,
                                                pw_upd2, upd_b2, ln_g, ln_b,
                                                agg_out, N_);
    }
}
